// GNN_39779987096058
// MI455X (gfx1250) — matrix operand order repaired and checked by reading, not compiled, not run on hardware
//
#include <hip/hip_runtime.h>
#include <hip/hip_bf16.h>
#include <math.h>

#define NB    64
#define NMEN  120
#define NENT  60
#define NSEN  18
#define NVIR  2
#define DENC  768
#define DTYPE 128
#define DD    896
#define NNODE 200
#define NTOT  (NB * NNODE)            /* 12800 */
#define NEDGE 409600
#define HTOT  ((long)NTOT * DD)       /* 11,468,800 */
#define ENTOUT ((long)NB * NENT * DD) /* 3,440,640 */

typedef __attribute__((ext_vector_type(16))) __bf16 v16bf;
typedef __attribute__((ext_vector_type(8)))  float  v8f;

union Frag { v16bf v; unsigned int u[8]; };

__device__ __forceinline__ unsigned short f2bf(float x) {
    unsigned int u = __float_as_uint(x);
    u += 0x7FFFu + ((u >> 16) & 1u);   // round-to-nearest-even
    return (unsigned short)(u >> 16);
}

__device__ __forceinline__ void wait_asynccnt0() {
#if defined(__has_builtin)
#if __has_builtin(__builtin_amdgcn_s_wait_asynccnt)
    __builtin_amdgcn_s_wait_asynccnt(0);
    return;
#endif
#endif
    asm volatile("s_wait_asynccnt 0x0" ::: "memory");
}

// async memory -> LDS, 16B per lane, tracked by ASYNCcnt (no VGPR round-trip)
__device__ __forceinline__ void async_b128_to_lds(unsigned lds_off, const void* gaddr) {
    asm volatile("global_load_async_to_lds_b128 %0, %1, off"
                 :: "v"(lds_off), "v"(gaddr) : "memory");
}

// ---------------- node assembly: concat hidden states + type embedding -----
__global__ void gnn_build_nodes(const float* __restrict__ men,
                                const float* __restrict__ ent,
                                const float* __restrict__ sen,
                                const float* __restrict__ vir,
                                const float* __restrict__ temb,
                                float* __restrict__ h, float* __restrict__ h0) {
    long i = (long)blockIdx.x * blockDim.x + threadIdx.x;
    if (i >= HTOT) return;
    int n = (int)(i / DD), f = (int)(i % DD);
    int b = n / NNODE, loc = n % NNODE;
    int t; const float* src;
    if (loc < NMEN)                { t = 0; src = men + ((long)b*NMEN + loc) * DENC; }
    else if (loc < NMEN+NENT)      { t = 1; src = ent + ((long)b*NENT + (loc-NMEN)) * DENC; }
    else if (loc < NMEN+NENT+NSEN) { t = 2; src = sen + ((long)b*NSEN + (loc-NMEN-NENT)) * DENC; }
    else                           { t = 3; src = vir + ((long)b*NVIR + (loc-NMEN-NENT-NSEN)) * DENC; }
    float v = (f < DENC) ? src[f] : temb[t*DTYPE + (f - DENC)];
    h[i] = v; h0[i] = v;
}

// ---------------- weights fp32 -> bf16 -------------------------------------
__global__ void gnn_conv_w(const float* __restrict__ w, unsigned short* __restrict__ wb, long n) {
    long i = (long)blockIdx.x * blockDim.x + threadIdx.x;
    if (i < n) wb[i] = f2bf(w[i]);
}

// ---------------- degree / dis ---------------------------------------------
__global__ void gnn_deg_init(float* deg) {
    int i = blockIdx.x * blockDim.x + threadIdx.x;
    if (i < NTOT) deg[i] = 1.0f;  // self loop
}
__global__ void gnn_deg_acc(const int* __restrict__ edst, float* deg) {
    int e = blockIdx.x * blockDim.x + threadIdx.x;
    if (e < NEDGE) unsafeAtomicAdd(&deg[edst[e]], 1.0f);
}
__global__ void gnn_dis(float* deg) {
    int i = blockIdx.x * blockDim.x + threadIdx.x;
    if (i < NTOT) deg[i] = rsqrtf(fmaxf(deg[i], 1.0f));
}

// ---------------- agg init with self-loop (no atomics) ----------------------
__global__ void gnn_selfloop(const float* __restrict__ h, const float* __restrict__ dis,
                             float* __restrict__ agg) {
    long i = (long)blockIdx.x * blockDim.x + threadIdx.x;
    if (i >= HTOT) return;
    float c = dis[i / DD];
    agg[i] = c * c * h[i];
}

// ---------------- edge scatter: agg[dst] += dis[s]*dis[d]*h[src] ------------
__global__ void gnn_spmm(const float* __restrict__ h, const float* __restrict__ dis,
                         const int* __restrict__ esrc, const int* __restrict__ edst,
                         float* __restrict__ agg) {
    long i = (long)blockIdx.x * blockDim.x + threadIdx.x;   // NEDGE * 224
    if (i >= (long)NEDGE * (DD/4)) return;
    int e = (int)(i / (DD/4));
    int c = (int)(i % (DD/4)) * 4;
    int s = esrc[e], d = edst[e];
    float coef = dis[s] * dis[d];
    const float4 hv = *(const float4*)(h + (long)s*DD + c);
    float* ap = agg + (long)d*DD + c;
    unsafeAtomicAdd(ap + 0, coef * hv.x);
    unsafeAtomicAdd(ap + 1, coef * hv.y);
    unsafeAtomicAdd(ap + 2, coef * hv.z);
    unsafeAtomicAdd(ap + 3, coef * hv.w);
}

// ---------------- support = 0.9*agg + 0.1*h0 (in place) + bf16 copy ---------
__global__ void gnn_support(const float* __restrict__ h0, float* __restrict__ agg,
                            unsigned short* __restrict__ abf) {
    long i = (long)blockIdx.x * blockDim.x + threadIdx.x;
    if (i >= HTOT) return;
    float s = 0.9f * agg[i] + 0.1f * h0[i];
    agg[i] = s;
    abf[i] = f2bf(s);
}

// ---------------- fused WMMA GEMM: h = relu((1-b)*supp + b*(supp @ W)) ------
#define BM 128
#define BN 128
#define BK 32
#define KSTEPS (DD / BK)   /* 28 */

__global__ __launch_bounds__(256)
void gnn_gemm_relu(const unsigned short* __restrict__ A,   // [NTOT][896] bf16 support
                   const unsigned short* __restrict__ Wb,  // [896][896] bf16 layer weight
                   const float* __restrict__ supp,         // [NTOT][896] f32 support
                   float* __restrict__ Hout,               // [NTOT][896]
                   float beta) {
    __shared__ __align__(16) unsigned short sA[2][BM * BK]; // row-major [row][k], dbl-buffered
    __shared__ __align__(16) unsigned short sB[2][BN * BK]; // col-major [n][k],   dbl-buffered

    const int t    = threadIdx.x;
    const int lane = t & 31;
    const int wave = t >> 5;
    const int half = lane >> 4;
    const int lr   = lane & 15;
    const int wm   = wave & 3;       // 4 waves along M -> 32 rows each
    const int wn   = wave >> 2;      // 2 waves along N -> 64 cols each
    const int bm   = blockIdx.x * BM;
    const int bn   = blockIdx.y * BN;

    // staging coordinates (round r adds 64 rows / 16 k-rows)
    const int arow = t >> 2,  ac8 = (t & 3) * 8;    // A: row, k-chunk
    const int bkr  = t >> 4,  bn8 = (t & 15) * 8;   // B: k-row, n-chunk

    v8f acc[2][4];
#pragma unroll
    for (int mi = 0; mi < 2; ++mi)
#pragma unroll
        for (int ni = 0; ni < 4; ++ni) acc[mi][ni] = (v8f)0.0f;

    uint4 breg[2];

    // --- stage A tile (async mem->LDS) for buffer `buf` at k-offset k0 ------
    auto stageA = [&](int buf, int k0) {
#pragma unroll
        for (int r = 0; r < 2; ++r) {
            int row = arow + 64 * r;
            unsigned lds = (unsigned)(size_t)(&sA[buf][row * BK + ac8]);
            async_b128_to_lds(lds, (const void*)(A + (long)(bm + row) * DD + k0 + ac8));
        }
    };
    // --- issue B tile global loads into registers ---------------------------
    auto loadB = [&](int k0) {
#pragma unroll
        for (int r = 0; r < 2; ++r)
            breg[r] = *(const uint4*)(Wb + (long)(k0 + bkr + 16 * r) * DD + bn + bn8);
    };
    // --- transpose-store B registers into LDS buffer ------------------------
    auto storeB = [&](int buf) {
#pragma unroll
        for (int r = 0; r < 2; ++r) {
            int kr = bkr + 16 * r;
            const unsigned short* p = (const unsigned short*)&breg[r];
#pragma unroll
            for (int q = 0; q < 8; ++q) sB[buf][(bn8 + q) * BK + kr] = p[q];
        }
    };
    // --- fragments + 8 WMMAs on buffer --------------------------------------
    auto compute = [&](int buf) {
        const unsigned short* pA = sA[buf];
        const unsigned short* pB = sB[buf];
        Frag a[2], b[4];
#pragma unroll
        for (int mi = 0; mi < 2; ++mi) {
            const unsigned short* base = pA + (wm * 32 + mi * 16 + lr) * BK;
#pragma unroll
            for (int j = 0; j < 8; ++j) {
                int kb = 2 * j + (j >= 4 ? 8 : 0) + 8 * half;   // A: K pair layout
                a[mi].u[j] = *(const unsigned int*)(base + kb);
            }
        }
#pragma unroll
        for (int ni = 0; ni < 4; ++ni) {
            const unsigned short* base = pB + (wn * 64 + ni * 16 + lr) * BK;
#pragma unroll
            for (int j = 0; j < 8; ++j)
                b[ni].u[j] = *(const unsigned int*)(base + 2 * j + (j >= 4 ? 8 : 0) + 8 * half);  // B: K pair layout
        }
#pragma unroll
        for (int mi = 0; mi < 2; ++mi)
#pragma unroll
            for (int ni = 0; ni < 4; ++ni)
                acc[mi][ni] = __builtin_amdgcn_wmma_f32_16x16x32_bf16(
                    false, a[mi].v, false, b[ni].v, (short)0, acc[mi][ni], false, false);
    };

    // --- software pipeline: stage k+1 while computing k ----------------------
    stageA(0, 0);
    loadB(0);
    storeB(0);
    wait_asynccnt0();
    __syncthreads();

    for (int kt = 0; kt < KSTEPS; ++kt) {
        const int cur  = kt & 1;
        const bool more = (kt + 1) < KSTEPS;
        if (more) {
            const int k0n = (kt + 1) * BK;
            stageA(1 - cur, k0n);          // async, overlaps with compute below
            loadB(k0n);                    // loads in flight during compute
            if (kt + 2 < KSTEPS)           // prefetch tile-after-next of W
                __builtin_prefetch(Wb + (long)((kt + 2) * BK + bkr) * DD + bn + bn8, 0, 0);
        }
        compute(cur);
        if (more) {
            storeB(1 - cur);               // latency hidden under the WMMAs above
            wait_asynccnt0();
        }
        __syncthreads();
    }

    // --- fused epilogue: relu((1-beta)*support + beta*gemm) ------------------
    const float omb = 1.0f - beta;
#pragma unroll
    for (int mi = 0; mi < 2; ++mi) {
#pragma unroll
        for (int ni = 0; ni < 4; ++ni) {
#pragma unroll
            for (int r = 0; r < 8; ++r) {
                int row = bm + wm * 32 + mi * 16 + 8 * half + r;  // C/D: M = r + 8*half
                int col = bn + wn * 64 + ni * 16 + lr;            //      N = lane%16
                long idx = (long)row * DD + col;
                Hout[idx] = fmaxf(0.0f, omb * supp[idx] + beta * acc[mi][ni][r]);
            }
        }
    }
}

// ---------------- outputs: (entity_out, out) concatenated -------------------
__global__ void gnn_write_out(const float* __restrict__ h, float* __restrict__ out) {
    long i = (long)blockIdx.x * blockDim.x + threadIdx.x;
    if (i >= ENTOUT + HTOT) return;
    if (i < ENTOUT) {
        int b = (int)(i / ((long)NENT * DD));
        int r = (int)(i % ((long)NENT * DD));
        int e = r / DD, f = r % DD;
        out[i] = h[((long)(b * NNODE + NMEN + e)) * DD + f];
    } else {
        out[i] = h[i - ENTOUT];
    }
}

extern "C" void kernel_launch(void* const* d_in, const int* in_sizes, int n_in,
                              void* d_out, int out_size, void* d_ws, size_t ws_size,
                              hipStream_t stream) {
    const float* men  = (const float*)d_in[0];
    const float* ent  = (const float*)d_in[1];
    const float* sen  = (const float*)d_in[2];
    const float* vir  = (const float*)d_in[3];
    const float* temb = (const float*)d_in[4];
    const float* gw   = (const float*)d_in[5];
    const int*   esrc = (const int*)d_in[6];
    const int*   edst = (const int*)d_in[7];
    float* out = (float*)d_out;

    size_t off = 0;
    auto carve = [&](size_t bytes) -> void* {
        void* p = (char*)d_ws + off;
        off += (bytes + 255) & ~(size_t)255;
        return p;
    };
    float*          h   = (float*)carve(HTOT * sizeof(float));
    float*          h0  = (float*)carve(HTOT * sizeof(float));
    float*          agg = (float*)carve(HTOT * sizeof(float));           // also "support"
    unsigned short* abf = (unsigned short*)carve(HTOT * sizeof(unsigned short));
    unsigned short* wbf = (unsigned short*)carve((size_t)4 * DD * DD * sizeof(unsigned short));
    float*          deg = (float*)carve(NTOT * sizeof(float));           // becomes dis
    (void)ws_size;

    const int T = 256;
    const int gElem  = (int)((HTOT + T - 1) / T);                  // 44800
    const long wN    = (long)4 * DD * DD;                          // 3,211,264
    const int gW     = (int)((wN + T - 1) / T);
    const int gEdgeF = (int)(((long)NEDGE * (DD/4) + T - 1) / T);  // 358400
    const int gOut   = (int)((ENTOUT + HTOT + T - 1) / T);

    gnn_build_nodes<<<gElem, T, 0, stream>>>(men, ent, sen, vir, temb, h, h0);
    gnn_conv_w<<<gW, T, 0, stream>>>(gw, wbf, wN);
    gnn_deg_init<<<(NTOT + T - 1) / T, T, 0, stream>>>(deg);
    gnn_deg_acc<<<(NEDGE + T - 1) / T, T, 0, stream>>>(edst, deg);
    gnn_dis<<<(NTOT + T - 1) / T, T, 0, stream>>>(deg);

    const float betas[4] = { logf(1.5f), logf(1.25f), logf(7.0f / 6.0f), logf(1.125f) };
    for (int l = 0; l < 4; ++l) {
        gnn_selfloop<<<gElem, T, 0, stream>>>(h, deg, agg);
        gnn_spmm<<<gEdgeF, T, 0, stream>>>(h, deg, esrc, edst, agg);
        gnn_support<<<gElem, T, 0, stream>>>(h0, agg, abf);
        dim3 grid(NTOT / BM, DD / BN);  // (100, 7)
        gnn_gemm_relu<<<grid, T, 0, stream>>>(abf, wbf + (size_t)l * DD * DD,
                                              agg, h, betas[l]);
    }
    gnn_write_out<<<gOut, T, 0, stream>>>(h, out);
}
